// DFlashAttention_90726889161320
// MI455X (gfx1250) — matrix operand order repaired and checked by reading, not compiled, not run on hardware
//
#include <hip/hip_runtime.h>
#include <stdint.h>

// ---------------- problem constants ----------------
#define HIDDEN    2048
#define N_HEADS   16
#define N_KV      8
#define HEAD_DIM  128
#define T_NOISE   512
#define T_CTX     1536
#define T_NEW     (T_CTX + T_NOISE)      // 2048
#define MAX_KV    8192
#define EPS_RMS   1e-6f
#define ROPE_TH   1000000.0f
#define SM_SCALE  0.08838834764831845f   // 1/sqrt(128)

typedef __attribute__((ext_vector_type(16))) __bf16 v16bf;
typedef __attribute__((ext_vector_type(2)))  __bf16 v2bf;
typedef __attribute__((ext_vector_type(8)))  float  v8f;

union FragB16 { uint4 q[2]; v16bf v; };
union PackBF  { v2bf v; uint32_t u; };

// Native conversions: compiler emits v_cvt_bf16_f32 / v_cvt_pk_bf16_f32.
__device__ __forceinline__ __bf16 f2bf(float f) { return (__bf16)f; }
__device__ __forceinline__ uint32_t pack2bf(float a, float b) {
  PackBF p; p.v.x = (__bf16)a; p.v.y = (__bf16)b; return p.u;
}

// =====================================================================
// Generic GEMM: C[M,N] = A[M,K] * B[K,N], fp32 in/out, bf16 WMMA compute.
// A rows < split come from A1, the rest from A2 (row-major both).
// Block = 128 threads (4 waves). Tile: 64(M) x 128(N) x 32(K),
// double-buffered LDS (one barrier per K-step, loads overlap WMMA).
// M % 64 == 0, N % 128 == 0, K % 32 == 0 (true for all uses here).
// =====================================================================
#define LDA_S 48   // bf16 elems per LDS A row (16B-aligned runs)
#define LDB_S 48

__device__ __forceinline__ void stage_A(const float* __restrict__ aptr, int kt,
                                        __bf16* __restrict__ lA, int tid) {
  const int arow = tid >> 1;    // 0..63
  const int aseg = tid & 1;     // 16-float segment
  const float4* s = (const float4*)(aptr + kt + aseg * 16);
  uint32_t* d = (uint32_t*)(&lA[arow * LDA_S + aseg * 16]);
  float4 v0 = s[0], v1 = s[1], v2 = s[2], v3 = s[3];
  d[0] = pack2bf(v0.x, v0.y); d[1] = pack2bf(v0.z, v0.w);
  d[2] = pack2bf(v1.x, v1.y); d[3] = pack2bf(v1.z, v1.w);
  d[4] = pack2bf(v2.x, v2.y); d[5] = pack2bf(v2.z, v2.w);
  d[6] = pack2bf(v3.x, v3.y); d[7] = pack2bf(v3.z, v3.w);
}

// B tile 32(K) x 128(N) -> lBt[col][k], each thread owns 4(K) x 8(cols),
// packing 4 K-values per column into one b64 store.
__device__ __forceinline__ void stage_B(const float* __restrict__ B, int N, int kt,
                                        int blockN, __bf16* __restrict__ lBt, int tid) {
  const int kg = tid >> 4;          // 0..7  -> k = kg*4 .. +3
  const int c0 = (tid & 15) * 8;    // 8 columns
  const float* src = B + (size_t)(kt + kg * 4) * N + blockN + c0;
  float r[4][8];
  #pragma unroll
  for (int rr = 0; rr < 4; ++rr) {
    const float4* s = (const float4*)(src + (size_t)rr * N);
    float4 a = s[0], b = s[1];
    r[rr][0] = a.x; r[rr][1] = a.y; r[rr][2] = a.z; r[rr][3] = a.w;
    r[rr][4] = b.x; r[rr][5] = b.y; r[rr][6] = b.z; r[rr][7] = b.w;
  }
  #pragma unroll
  for (int i = 0; i < 8; ++i) {
    uint2 u;
    u.x = pack2bf(r[0][i], r[1][i]);
    u.y = pack2bf(r[2][i], r[3][i]);
    *(uint2*)(&lBt[(c0 + i) * LDB_S + kg * 4]) = u;
  }
}

__global__ __launch_bounds__(128)
void gemm_bf16_wmma(const float* __restrict__ A1, const float* __restrict__ A2,
                    int split, const float* __restrict__ B,
                    float* __restrict__ C, int M, int N, int K) {
  __shared__ __align__(16) __bf16 lA[2][64 * LDA_S];
  __shared__ __align__(16) __bf16 lB[2][128 * LDB_S];

  const int tid  = threadIdx.x;
  const int wave = tid >> 5;
  const int lane = tid & 31;
  const int n    = lane & 15;
  const int g    = lane >> 4;

  const int blockM = blockIdx.x * 64;
  const int blockN = blockIdx.y * 128;

  v8f acc[8];
  v8f zero = {};
  #pragma unroll
  for (int c = 0; c < 8; ++c) acc[c] = zero;

  const int arow = tid >> 1;
  const int grow = blockM + arow;
  const float* aptr = (grow < split) ? (A1 + (size_t)grow * K)
                                     : (A2 + (size_t)(grow - split) * K);

  const int NT = K / 32;
  stage_A(aptr, 0, lA[0], tid);
  stage_B(B, N, 0, blockN, lB[0], tid);
  __syncthreads();

  for (int it = 0; it < NT; ++it) {
    __bf16* curA = lA[it & 1];
    __bf16* curB = lB[it & 1];
    if (it + 1 < NT) {
      stage_A(aptr, (it + 1) * 32, lA[(it + 1) & 1], tid);
      stage_B(B, N, (it + 1) * 32, blockN, lB[(it + 1) & 1], tid);
      if (it + 2 < NT) __builtin_prefetch((const void*)(aptr + (it + 2) * 32), 0, 1);
    }

    FragB16 fa;
    const __bf16* arp = &curA[(wave * 16 + n) * LDA_S];
    fa.q[0] = *(const uint4*)(arp + g * 16);
    fa.q[1] = *(const uint4*)(arp + g * 16 + 8);
    #pragma unroll
    for (int c = 0; c < 8; ++c) {
      FragB16 fb;
      const __bf16* bp = &curB[(c * 16 + n) * LDB_S + g * 16];
      fb.q[0] = *(const uint4*)(bp);
      fb.q[1] = *(const uint4*)(bp + 8);
      acc[c] = __builtin_amdgcn_wmma_f32_16x16x32_bf16(false, fa.v, false, fb.v,
                                                       (short)0, acc[c], false, false);
    }
    __syncthreads();
  }

  // ---- epilogue: C layout VGPR j -> M = j + 8*g, N = lane&15 ----
  const int crow0 = blockM + wave * 16 + g * 8;
  #pragma unroll
  for (int c = 0; c < 8; ++c) {
    #pragma unroll
    for (int j = 0; j < 8; ++j)
      C[(size_t)(crow0 + j) * N + blockN + c * 16 + n] = acc[c][j];
  }
}

// =====================================================================
// KV-cache pass-through copy (streams at HBM rate).
// =====================================================================
__global__ void copy_f4(const float4* __restrict__ src, float4* __restrict__ dst, int n4) {
  int i = blockIdx.x * blockDim.x + threadIdx.x;
  int stride = gridDim.x * blockDim.x;
  for (; i < n4; i += stride) dst[i] = src[i];
}

// =====================================================================
// Q: per-(token,head) RMSNorm + RoPE, output bf16 for attention.
// grid (T_NOISE, N_HEADS), block 128 (one thread per head-dim element).
// =====================================================================
__global__ __launch_bounds__(128)
void q_norm_rope(const float* __restrict__ q_raw, const float* __restrict__ q_scale,
                 const int* __restrict__ pos, uint16_t* __restrict__ q_bf) {
  const int t = blockIdx.x, h = blockIdx.y, d = threadIdx.x;
  __shared__ float red[4];
  __shared__ float nrm[HEAD_DIM];

  const size_t base = ((size_t)t * N_HEADS + h) * HEAD_DIM;
  float x = q_raw[base + d];
  float ss = x * x;
  #pragma unroll
  for (int m = 16; m >= 1; m >>= 1) ss += __shfl_xor(ss, m, 32);
  if ((threadIdx.x & 31) == 0) red[threadIdx.x >> 5] = ss;
  __syncthreads();
  float tot = red[0] + red[1] + red[2] + red[3];
  float r = rsqrtf(tot * (1.0f / HEAD_DIM) + EPS_RMS);
  nrm[d] = x * r * q_scale[d];
  __syncthreads();

  float p = (float)pos[t];
  int i = d & 63;
  float invf = __expf(-(2.0f * (float)i / (float)HEAD_DIM) * __logf(ROPE_TH));
  float s, c; __sincosf(p * invf, &s, &c);
  float out = (d < 64) ? (nrm[d] * c - nrm[d + 64] * s)
                       : (nrm[d] * c + nrm[d - 64] * s);
  PackBF pk; pk.v.x = (__bf16)out; pk.v.y = (__bf16)0.0f;
  q_bf[base + d] = (uint16_t)(pk.u & 0xFFFFu);
}

// =====================================================================
// K/V: RMSNorm(K)+RoPE(K), then GQA-repeat scatter into cache outputs at
// dynamic cache_len (read from device). grid (T_NEW, N_KV), block 128.
// =====================================================================
__global__ __launch_bounds__(128)
void kv_norm_rope_scatter(const float* __restrict__ k_raw, const float* __restrict__ v_raw,
                          const float* __restrict__ k_scale,
                          const int* __restrict__ ctx_pos, const int* __restrict__ noise_pos,
                          const int* __restrict__ cache_len_p,
                          float* __restrict__ kout, float* __restrict__ vout) {
  const int t = blockIdx.x, kh = blockIdx.y, d = threadIdx.x;
  __shared__ float red[4];
  __shared__ float nrm[HEAD_DIM];

  const size_t base = ((size_t)t * N_KV + kh) * HEAD_DIM;
  float x = k_raw[base + d];
  float ss = x * x;
  #pragma unroll
  for (int m = 16; m >= 1; m >>= 1) ss += __shfl_xor(ss, m, 32);
  if ((threadIdx.x & 31) == 0) red[threadIdx.x >> 5] = ss;
  __syncthreads();
  float tot = red[0] + red[1] + red[2] + red[3];
  float r = rsqrtf(tot * (1.0f / HEAD_DIM) + EPS_RMS);
  nrm[d] = x * r * k_scale[d];
  __syncthreads();

  int p = (t < T_CTX) ? ctx_pos[t] : noise_pos[t - T_CTX];
  int i = d & 63;
  float invf = __expf(-(2.0f * (float)i / (float)HEAD_DIM) * __logf(ROPE_TH));
  float s, c; __sincosf((float)p * invf, &s, &c);
  float kval = (d < 64) ? (nrm[d] * c - nrm[d + 64] * s)
                        : (nrm[d] * c + nrm[d - 64] * s);
  float vval = v_raw[base + d];

  int row = cache_len_p[0] + t;
  size_t o0 = ((size_t)(2 * kh)     * MAX_KV + row) * HEAD_DIM + d;
  size_t o1 = ((size_t)(2 * kh + 1) * MAX_KV + row) * HEAD_DIM + d;
  kout[o0] = kval; kout[o1] = kval;
  vout[o0] = vval; vout[o1] = vval;
}

// =====================================================================
// Flash attention: grid (T_NOISE/16, N_HEADS), block 128 = 4 waves.
// Each wave streams interleaved 32-key chunks with online softmax,
// QK^T and P*V both via v_wmma_f32_16x16x32_bf16. Cross-wave LSE merge.
// =====================================================================
#define AW 4  // waves per block

__global__ __launch_bounds__(128)
void attn_fa(const uint16_t* __restrict__ qbf_u, const float* __restrict__ kc,
             const float* __restrict__ vc, const int* __restrict__ cache_len_p,
             float* __restrict__ attn) {
  const int qb = blockIdx.x;
  const int h  = blockIdx.y;
  const int tid = threadIdx.x;
  const int w = tid >> 5, lane = tid & 31;
  const int n = lane & 15, g = lane >> 4;

  __shared__ __align__(16) __bf16 pP[AW][16 * 48];
  __shared__ __align__(16) __bf16 vT[AW][HEAD_DIM * 32];  // [dim][key]
  __shared__ float outAcc[16][HEAD_DIM];
  __shared__ float mArr[AW][16];
  __shared__ float sArr[AW][16];

  for (int i = tid; i < 16 * HEAD_DIM; i += 128) ((float*)outAcc)[i] = 0.0f;
  __syncthreads();

  const int len = cache_len_p[0] + T_NEW;
  const __bf16* qbf = (const __bf16*)qbf_u;

  // Q fragments: 4 chunks of 16x32 (A operand in this code's contiguous k order: lane = M + 16*Khalf)
  FragB16 qf[4];
  {
    const __bf16* qrow = qbf + ((size_t)(qb * 16 + n) * N_HEADS + h) * HEAD_DIM;
    #pragma unroll
    for (int kch = 0; kch < 4; ++kch) {
      qf[kch].q[0] = *(const uint4*)(qrow + kch * 32 + g * 16);
      qf[kch].q[1] = *(const uint4*)(qrow + kch * 32 + g * 16 + 8);
    }
  }

  v8f zero = {};
  v8f acc[8];
  #pragma unroll
  for (int d = 0; d < 8; ++d) acc[d] = zero;
  float rm[8], rs[8];
  #pragma unroll
  for (int j = 0; j < 8; ++j) { rm[j] = -3.0e38f; rs[j] = 0.0f; }

  const float* kbase = kc + (size_t)h * MAX_KV * HEAD_DIM;
  const float* vbase = vc + (size_t)h * MAX_KV * HEAD_DIM;

  for (int s0 = w * 32; s0 < len; s0 += AW * 32) {
    // ---- scores: two 16x16 tiles over 4 K-dim chunks ----
    v8f st[2]; st[0] = zero; st[1] = zero;
    #pragma unroll
    for (int c = 0; c < 2; ++c) {
      int key = s0 + c * 16 + n;
      int kcl = key < (MAX_KV - 1) ? key : (MAX_KV - 1);
      const float* krow = kbase + (size_t)kcl * HEAD_DIM + g * 16;
      #pragma unroll
      for (int kch = 0; kch < 4; ++kch) {
        const float4* f4 = (const float4*)(krow + kch * 32);
        float4 a = f4[0], b = f4[1], e = f4[2], f = f4[3];
        FragB16 kf;
        uint32_t* u = (uint32_t*)&kf;
        u[0] = pack2bf(a.x, a.y); u[1] = pack2bf(a.z, a.w);
        u[2] = pack2bf(b.x, b.y); u[3] = pack2bf(b.z, b.w);
        u[4] = pack2bf(e.x, e.y); u[5] = pack2bf(e.z, e.w);
        u[6] = pack2bf(f.x, f.y); u[7] = pack2bf(f.z, f.w);
        st[c] = __builtin_amdgcn_wmma_f32_16x16x32_bf16(false, qf[kch].v, false, kf.v,
                                                        (short)0, st[c], false, false);
      }
    }

    // ---- stage V^T (per-wave): lane owns one key row ----
    {
      int key = s0 + lane;
      int kcl = key < (MAX_KV - 1) ? key : (MAX_KV - 1);
      const float4* vrow = (const float4*)(vbase + (size_t)kcl * HEAD_DIM);
      __bf16* vt = vT[w];
      #pragma unroll
      for (int q4 = 0; q4 < 32; ++q4) {
        float4 v = vrow[q4];
        vt[(q4 * 4 + 0) * 32 + lane] = f2bf(v.x);
        vt[(q4 * 4 + 1) * 32 + lane] = f2bf(v.y);
        vt[(q4 * 4 + 2) * 32 + lane] = f2bf(v.z);
        vt[(q4 * 4 + 3) * 32 + lane] = f2bf(v.w);
      }
    }

    // ---- online softmax (rows striped: row = j + 8*g, col = lane&15) ----
    float p0[8], p1[8];
    #pragma unroll
    for (int j = 0; j < 8; ++j) {
      float a = st[0][j] * SM_SCALE;
      float b = st[1][j] * SM_SCALE;
      if (s0 + n >= len)      a = -3.0e38f;
      if (s0 + 16 + n >= len) b = -3.0e38f;
      float t = fmaxf(a, b);
      t = fmaxf(t, __shfl_xor(t, 1, 32));
      t = fmaxf(t, __shfl_xor(t, 2, 32));
      t = fmaxf(t, __shfl_xor(t, 4, 32));
      t = fmaxf(t, __shfl_xor(t, 8, 32));
      float mnew = fmaxf(rm[j], t);
      float scale = __expf(rm[j] - mnew);
      float e0 = __expf(a - mnew);
      float e1 = __expf(b - mnew);
      float sum = e0 + e1;
      sum += __shfl_xor(sum, 1, 32);
      sum += __shfl_xor(sum, 2, 32);
      sum += __shfl_xor(sum, 4, 32);
      sum += __shfl_xor(sum, 8, 32);
      rs[j] = rs[j] * scale + sum;
      rm[j] = mnew;
      p0[j] = e0; p1[j] = e1;
      #pragma unroll
      for (int d = 0; d < 8; ++d) acc[d][j] *= scale;
    }

    // ---- P -> LDS (C layout -> A layout round-trip) ----
    {
      __bf16* pp = pP[w];
      #pragma unroll
      for (int j = 0; j < 8; ++j) {
        int row = j + 8 * g;
        uint32_t pr = pack2bf(p0[j], p1[j]);
        pp[row * 48 + n]      = ((PackBF*)&pr)->v.x;
        pp[row * 48 + 16 + n] = ((PackBF*)&pr)->v.y;
      }
    }

    // ---- P(16x32) * V(32x128): 8 WMMAs ----
    {
      FragB16 ap;
      const __bf16* prow = &pP[w][n * 48];
      ap.q[0] = *(const uint4*)(prow + g * 16);
      ap.q[1] = *(const uint4*)(prow + g * 16 + 8);
      const __bf16* vt = vT[w];
      #pragma unroll
      for (int d = 0; d < 8; ++d) {
        FragB16 vb;
        const __bf16* vp = vt + (size_t)(d * 16 + n) * 32 + g * 16;
        vb.q[0] = *(const uint4*)(vp);
        vb.q[1] = *(const uint4*)(vp + 8);
        acc[d] = __builtin_amdgcn_wmma_f32_16x16x32_bf16(false, ap.v, false, vb.v,
                                                         (short)0, acc[d], false, false);
      }
    }
  }

  // ---- cross-wave LSE merge ----
  if (n == 0) {
    #pragma unroll
    for (int j = 0; j < 8; ++j) {
      mArr[w][j + 8 * g] = rm[j];
      sArr[w][j + 8 * g] = rs[j];
    }
  }
  __syncthreads();

  #pragma unroll
  for (int j = 0; j < 8; ++j) {
    int row = j + 8 * g;
    float M = mArr[0][row];
    for (int ww = 1; ww < AW; ++ww) M = fmaxf(M, mArr[ww][row]);
    float fac = __expf(rm[j] - M);
    #pragma unroll
    for (int d = 0; d < 8; ++d)
      atomicAdd(&outAcc[row][d * 16 + n], acc[d][j] * fac);
  }
  __syncthreads();

  {
    int row = tid >> 3;
    int seg = tid & 7;
    float M = mArr[0][row];
    for (int ww = 1; ww < AW; ++ww) M = fmaxf(M, mArr[ww][row]);
    float denom = 0.0f;
    for (int ww = 0; ww < AW; ++ww) denom += sArr[ww][row] * __expf(mArr[ww][row] - M);
    float inv = 1.0f / denom;
    float* dst = attn + ((size_t)(qb * 16 + row) * N_HEADS + h) * HEAD_DIM + seg * 16;
    #pragma unroll
    for (int i = 0; i < 16; ++i) dst[i] = outAcc[row][seg * 16 + i] * inv;
  }
}

// =====================================================================
// Host launcher
// =====================================================================
extern "C" void kernel_launch(void* const* d_in, const int* in_sizes, int n_in,
                              void* d_out, int out_size, void* d_ws, size_t ws_size,
                              hipStream_t stream) {
  const float* x_noise   = (const float*)d_in[0];
  const float* tgt_hid   = (const float*)d_in[1];
  const int*   noise_pos = (const int*)d_in[2];
  const int*   ctx_pos   = (const int*)d_in[3];
  const float* kc_in     = (const float*)d_in[4];
  const float* vc_in     = (const float*)d_in[5];
  const int*   cache_len = (const int*)d_in[6];
  const float* wq        = (const float*)d_in[7];
  const float* wk        = (const float*)d_in[8];
  const float* wv        = (const float*)d_in[9];
  const float* wo        = (const float*)d_in[10];
  const float* q_scale   = (const float*)d_in[11];
  const float* k_scale   = (const float*)d_in[12];

  float* out      = (float*)d_out;
  float* out_main = out;                               // 512*2048
  float* out_k    = out + (size_t)T_NOISE * HIDDEN;    // 16*8192*128
  float* out_v    = out_k + (size_t)N_HEADS * MAX_KV * HEAD_DIM;

  char* ws = (char*)d_ws;
  float*    q_raw = (float*)(ws + 0);                         // 512*2048 f32
  float*    k_raw = (float*)(ws + 4194304);                   // 2048*1024 f32
  float*    v_raw = (float*)(ws + 12582912);                  // 2048*1024 f32
  float*    attn  = (float*)(ws + 20971520);                  // 512*2048 f32
  uint16_t* q_bf  = (uint16_t*)(ws + 25165824);               // 512*2048 bf16

  // 1) projections (bf16 WMMA GEMMs)
  gemm_bf16_wmma<<<dim3(T_NOISE / 64, HIDDEN / 128), 128, 0, stream>>>(
      x_noise, x_noise, T_NOISE, wq, q_raw, T_NOISE, HIDDEN, HIDDEN);
  gemm_bf16_wmma<<<dim3(T_NEW / 64, (N_KV * HEAD_DIM) / 128), 128, 0, stream>>>(
      tgt_hid, x_noise, T_CTX, wk, k_raw, T_NEW, N_KV * HEAD_DIM, HIDDEN);
  gemm_bf16_wmma<<<dim3(T_NEW / 64, (N_KV * HEAD_DIM) / 128), 128, 0, stream>>>(
      tgt_hid, x_noise, T_CTX, wv, v_raw, T_NEW, N_KV * HEAD_DIM, HIDDEN);

  // 2) cache pass-through
  const int cache_f4 = (N_HEADS * MAX_KV * HEAD_DIM) / 4;
  copy_f4<<<2048, 256, 0, stream>>>((const float4*)kc_in, (float4*)out_k, cache_f4);
  copy_f4<<<2048, 256, 0, stream>>>((const float4*)vc_in, (float4*)out_v, cache_f4);

  // 3) norm + rope (+ scatter)
  q_norm_rope<<<dim3(T_NOISE, N_HEADS), 128, 0, stream>>>(q_raw, q_scale, noise_pos, q_bf);
  kv_norm_rope_scatter<<<dim3(T_NEW, N_KV), 128, 0, stream>>>(
      k_raw, v_raw, k_scale, ctx_pos, noise_pos, cache_len, out_k, out_v);

  // 4) attention
  attn_fa<<<dim3(T_NOISE / 16, N_HEADS), 128, 0, stream>>>(q_bf, out_k, out_v, cache_len, attn);

  // 5) output projection
  gemm_bf16_wmma<<<dim3(T_NOISE / 64, HIDDEN / 128), 128, 0, stream>>>(
      attn, attn, T_NOISE, wo, out_main, T_NOISE, HIDDEN, HIDDEN);
}
